// PointNetPP_35296041239308
// MI455X (gfx1250) — compile-verified, matrix operand order checked by reading, not run on hardware
//
#include <hip/hip_runtime.h>
#include <math.h>

#define BB 4
#define NPTS 16384

typedef __attribute__((ext_vector_type(16))) _Float16 v16h;
typedef __attribute__((ext_vector_type(8)))  _Float16 v8h;
typedef __attribute__((ext_vector_type(8)))  float    v8f;

// ---------------- l0p = [features(3), xyz(3)] as f16 ----------------
__global__ void k_build_l0p(const float* __restrict__ feat, const float* __restrict__ xyz,
                            _Float16* __restrict__ l0p, int total) {
  int i = blockIdx.x * blockDim.x + threadIdx.x;
  if (i >= total) return;
  int c = i % 6; int pt = i / 6;
  float v = (c < 3) ? feat[pt * 3 + c] : xyz[pt * 3 + (c - 3)];
  l0p[i] = (_Float16)v;
}

// ---------------- weight pack: f32 (Cout,K) -> f16 (Cout,Kpad) zero-padded -----------
__global__ void k_packW(const float* __restrict__ W, _Float16* __restrict__ Wh,
                        int K, int Kpad, long total) {
  long i = (long)blockIdx.x * blockDim.x + threadIdx.x;
  if (i >= total) return;
  int k = (int)(i % Kpad); long oc = i / Kpad;
  Wh[i] = (k < K) ? (_Float16)W[oc * K + k] : (_Float16)0.0f;
}

// ---------------- farthest point sampling (one block per batch) ----------------
__global__ void k_fps(const float* __restrict__ xyz, float* __restrict__ mind,
                      int N, int npoint, int* __restrict__ sidx) {
  int b = blockIdx.x;
  const float* p = xyz + (long)b * N * 3;
  float* md = mind + (long)b * N;
  __shared__ float sv[256];
  __shared__ int   si[256];
  __shared__ int   s_last;
  int t = threadIdx.x;
  for (int i = t; i < N; i += 256) md[i] = 1e10f;
  if (t == 0) { sidx[(long)b * npoint] = 0; s_last = 0; }
  __syncthreads();
  for (int it = 1; it < npoint; ++it) {
    int last = s_last;
    float lx = p[last*3+0], ly = p[last*3+1], lz = p[last*3+2];
    float best = -1.0f; int bi = 0;
    for (int i = t; i < N; i += 256) {
      float dx = p[i*3+0]-lx, dy = p[i*3+1]-ly, dz = p[i*3+2]-lz;
      float d = dx*dx + dy*dy + dz*dz;
      float m = md[i]; m = fminf(m, d); md[i] = m;
      if (m > best) { best = m; bi = i; }
    }
    sv[t] = best; si[t] = bi;
    __syncthreads();
    for (int s = 128; s > 0; s >>= 1) {
      if (t < s && sv[t+s] > sv[t]) { sv[t] = sv[t+s]; si[t] = si[t+s]; }
      __syncthreads();
    }
    if (t == 0) { sidx[(long)b * npoint + it] = si[0]; s_last = si[0]; }
    __syncthreads();
  }
}

__global__ void k_gather_xyz(const float* __restrict__ src, const int* __restrict__ sidx,
                             int S, int Nsrc, float* __restrict__ out, int total) {
  int i = blockIdx.x * blockDim.x + threadIdx.x;
  if (i >= total) return;
  int b = i / S;
  int id = sidx[i];
  const float* s = src + ((long)b * Nsrc + id) * 3;
  out[i*3+0] = s[0]; out[i*3+1] = s[1]; out[i*3+2] = s[2];
}

// ---------------- brute-force kNN (k=32) with radius mask ----------------
__global__ void k_knn(const float* __restrict__ q, const float* __restrict__ p,
                      int S, int N, float r2, int* __restrict__ out) {
  int i = blockIdx.x * blockDim.x + threadIdx.x;
  if (i >= BB * S) return;
  int b = i / S;
  float qx = q[i*3], qy = q[i*3+1], qz = q[i*3+2];
  const float* pb = p + (long)b * N * 3;
  float bd[32]; int bid[32];
  #pragma unroll
  for (int j = 0; j < 32; j++) { bd[j] = 3e38f; bid[j] = 0; }
  for (int n = 0; n < N; n++) {
    float dx = pb[n*3]-qx, dy = pb[n*3+1]-qy, dz = pb[n*3+2]-qz;
    float d = dx*dx + dy*dy + dz*dz;
    if (d < bd[31]) {
      int j = 31;
      while (j > 0 && bd[j-1] > d) { bd[j] = bd[j-1]; bid[j] = bid[j-1]; j--; }
      bd[j] = d; bid[j] = n;
    }
  }
  #pragma unroll
  for (int j = 0; j < 32; j++) out[(long)i*32 + j] = (bd[j] < r2) ? bid[j] : 0;
}

// ---- group: rows=(b,s,kk), cols=[dxyz(3), feat(C)], f16 out, zero-padded to Kpad ----
__global__ void k_group(const float* __restrict__ xyz, const _Float16* __restrict__ feat,
                        const float* __restrict__ nxyz, const int* __restrict__ idx,
                        int S, int N, int C, int Kpad, _Float16* __restrict__ X, long total) {
  long i = (long)blockIdx.x * blockDim.x + threadIdx.x;
  if (i >= total) return;                       // total = rows * Kpad
  int c = (int)(i % Kpad); long row = i / Kpad;
  int kk = (int)(row & 31); long bs = row >> 5;
  int b = (int)(bs / S);
  int id = idx[bs * 32 + kk];
  _Float16 v;
  if (c < 3)          v = (_Float16)(xyz[((long)b * N + id) * 3 + c] - nxyz[bs * 3 + c]);
  else if (c < 3 + C) v = feat[((long)b * N + id) * C + (c - 3)];
  else                v = (_Float16)0.0f;
  X[i] = v;
}

// ---------------- WMMA GEMM: branchless, b128 loads, 4 waves/block ----------------
// mode 0: folded BN + ReLU -> f16 Yh ; mode 1: +bias -> f32 Yf
__global__ void k_gemm_wmma(const _Float16* __restrict__ X, const _Float16* __restrict__ W,
                            const float* __restrict__ bias, const float* __restrict__ gam,
                            const float* __restrict__ bet, const float* __restrict__ mu,
                            const float* __restrict__ var,
                            _Float16* __restrict__ Yh, float* __restrict__ Yf,
                            int Kpad, int Cout, int mode) {
  int lane = threadIdx.x & 31;
  int wv   = threadIdx.x >> 5;                 // 4 waves, 4 M-tiles per block
  int row0 = (blockIdx.x * 4 + wv) << 4;
  int col0 = blockIdx.y << 4;
  int r  = lane & 15;
  int hi = lane >> 4;                          // lanes 16-31: K offsets +8 / +24
  const _Float16* xr = X + (size_t)(row0 + r) * Kpad + (hi << 3);
  const _Float16* wr = W + (size_t)(col0 + r) * Kpad + (hi << 3);
  v8f acc = {};
  for (int k0 = 0; k0 < Kpad; k0 += 32) {
    v8h a0 = *(const v8h*)(xr + k0);
    v8h a1 = *(const v8h*)(xr + k0 + 16);
    v8h b0 = *(const v8h*)(wr + k0);
    v8h b1 = *(const v8h*)(wr + k0 + 16);
    v16h a, b;
    #pragma unroll
    for (int e = 0; e < 8; e++) {
      a[e] = a0[e]; a[e + 8] = a1[e];
      b[e] = b0[e]; b[e + 8] = b1[e];
    }
    acc = __builtin_amdgcn_wmma_f32_16x16x32_f16(false, a, false, b,
                                                 (short)0, acc, false, false);
  }
  int col = col0 + r;
  float sc = 1.0f, sb;
  if (mode == 0) {
    sc = gam[col] * rsqrtf(var[col] + 1e-5f);
    sb = (bias[col] - mu[col]) * sc + bet[col];
  } else {
    sb = bias[col];
  }
  #pragma unroll
  for (int rr = 0; rr < 8; rr++) {
    int ro = row0 + rr + (hi << 3);            // C/D: lanes 16-31 hold rows M+8
    float v = acc[rr] * sc + sb;
    if (mode == 0) Yh[(size_t)ro * Cout + col] = (_Float16)fmaxf(v, 0.0f);
    else           Yf[(size_t)ro * Cout + col] = v;
  }
}

// ---------------- max over 32 neighbors (f16 in/out) ----------------
__global__ void k_maxpool(const _Float16* __restrict__ X, int C,
                          _Float16* __restrict__ Y, long total) {
  long i = (long)blockIdx.x * blockDim.x + threadIdx.x;
  if (i >= total) return;
  int c = (int)(i % C); long g = i / C;
  float mx = -3e38f;
  for (int kk = 0; kk < 32; kk++) mx = fmaxf(mx, (float)X[((g << 5) + kk) * C + c]);
  Y[i] = (_Float16)mx;
}

// ------- 3-NN inverse-distance interp + concat, f16 out, zero-padded to Kpad -------
__global__ void k_fp_interp(const float* __restrict__ x1, const float* __restrict__ x2,
                            const _Float16* __restrict__ f2, const _Float16* __restrict__ f1,
                            int N1, int N2, int C2, int C1, int Kpad,
                            _Float16* __restrict__ X) {
  int i = blockIdx.x * blockDim.x + threadIdx.x;
  if (i >= BB * N1) return;
  int b = i / N1;
  float qx = x1[i*3], qy = x1[i*3+1], qz = x1[i*3+2];
  const float* pb = x2 + (long)b * N2 * 3;
  float d0 = 3e38f, d1 = 3e38f, d2 = 3e38f; int i0 = 0, i1 = 0, i2 = 0;
  for (int n = 0; n < N2; n++) {
    float dx = pb[n*3]-qx, dy = pb[n*3+1]-qy, dz = pb[n*3+2]-qz;
    float d = dx*dx + dy*dy + dz*dz;
    if (d < d0)      { d2 = d1; i2 = i1; d1 = d0; i1 = i0; d0 = d; i0 = n; }
    else if (d < d1) { d2 = d1; i2 = i1; d1 = d;  i1 = n; }
    else if (d < d2) { d2 = d;  i2 = n; }
  }
  float w0 = 1.0f / (sqrtf(fmaxf(d0, 0.0f)) + 1e-10f);
  float w1 = 1.0f / (sqrtf(fmaxf(d1, 0.0f)) + 1e-10f);
  float w2 = 1.0f / (sqrtf(fmaxf(d2, 0.0f)) + 1e-10f);
  float ws = w0 + w1 + w2; w0 /= ws; w1 /= ws; w2 /= ws;
  const _Float16* r0 = f2 + ((long)b * N2 + i0) * C2;
  const _Float16* r1 = f2 + ((long)b * N2 + i1) * C2;
  const _Float16* r2p = f2 + ((long)b * N2 + i2) * C2;
  _Float16* Xr = X + (long)i * Kpad;
  for (int c = 0; c < C2; c++)
    Xr[c] = (_Float16)(w0 * (float)r0[c] + w1 * (float)r1[c] + w2 * (float)r2p[c]);
  const _Float16* fr = f1 + (long)i * C1;
  for (int c = 0; c < C1; c++) Xr[C2 + c] = fr[c];
  for (int c = C2 + C1; c < Kpad; c++) Xr[c] = (_Float16)0.0f;
}

// ---------------- L2-normalize over channels + transpose to (B,128,N) ----------------
__global__ void k_norm_out(const float* __restrict__ X, float* __restrict__ out, int Nn) {
  int i = blockIdx.x * blockDim.x + threadIdx.x;
  if (i >= BB * Nn) return;
  int b = i / Nn; int n = i % Nn;
  const float* xr = X + (long)i * 128;
  float s = 0.0f;
  for (int o = 0; o < 128; o++) { float v = xr[o]; s += v * v; }
  float inv = 1.0f / fmaxf(sqrtf(s), 1e-12f);
  for (int o = 0; o < 128; o++) out[((long)b * 128 + o) * Nn + n] = xr[o] * inv;
}

struct LayerP { const float *W, *b, *g, *be, *m, *v; };

extern "C" void kernel_launch(void* const* d_in, const int* in_sizes, int n_in,
                              void* d_out, int out_size, void* d_ws, size_t ws_size,
                              hipStream_t stream) {
  (void)in_sizes; (void)n_in; (void)out_size; (void)ws_size;
  const float* xyz  = (const float*)d_in[0];
  const float* feat = (const float*)d_in[1];
  int p = 2;
  auto getL = [&]() { LayerP L{ (const float*)d_in[p],   (const float*)d_in[p+1],
                                (const float*)d_in[p+2], (const float*)d_in[p+3],
                                (const float*)d_in[p+4], (const float*)d_in[p+5] };
                      p += 6; return L; };
  LayerP sa1[3] = { getL(), getL(), getL() };
  LayerP sa2[3] = { getL(), getL(), getL() };
  LayerP sa3[3] = { getL(), getL(), getL() };
  LayerP fp3[2] = { getL(), getL() };
  LayerP fp2[2] = { getL(), getL() };
  LayerP fp1[2] = { getL(), getL() };
  LayerP cls1   = getL();
  LayerP cls2{ (const float*)d_in[p], (const float*)d_in[p+1],
               nullptr, nullptr, nullptr, nullptr };

  // byte bump-allocator on d_ws, 256B aligned chunks
  char* base = (char*)d_ws;
  size_t off = 0;
  auto AB = [&](size_t bytes) { void* r = base + off; off += (bytes + 255) & ~(size_t)255; return r; };

  float*    mind  = (float*)   AB((size_t)BB * NPTS * 4);
  int*      sidx1 = (int*)     AB(BB * 1024 * 4);
  float*    nx1   = (float*)   AB(BB * 1024 * 3 * 4);
  int*      knn1  = (int*)     AB(BB * 1024 * 32 * 4);
  int*      sidx2 = (int*)     AB(BB * 256 * 4);
  float*    nx2   = (float*)   AB(BB * 256 * 3 * 4);
  int*      knn2  = (int*)     AB(BB * 256 * 32 * 4);
  int*      sidx3 = (int*)     AB(BB * 64 * 4);
  float*    nx3   = (float*)   AB(BB * 64 * 3 * 4);
  int*      knn3  = (int*)     AB(BB * 64 * 32 * 4);
  _Float16* l0p   = (_Float16*)AB((size_t)BB * NPTS * 6 * 2);
  _Float16* l1p   = (_Float16*)AB((size_t)BB * 1024 * 128 * 2);
  _Float16* l2p   = (_Float16*)AB((size_t)BB * 256 * 256 * 2);
  _Float16* l3p   = (_Float16*)AB((size_t)BB * 64 * 512 * 2);
  _Float16* fp3o  = (_Float16*)AB((size_t)BB * 256 * 256 * 2);
  _Float16* fp2o  = (_Float16*)AB((size_t)BB * 1024 * 128 * 2);
  _Float16* bufA  = (_Float16*)AB((size_t)131072 * 128 * 2);
  _Float16* bufB  = (_Float16*)AB((size_t)131072 * 128 * 2);
  float*    clsf  = (float*)   AB((size_t)65536 * 128 * 4);

  auto cdiv = [](long a, long b) { return (int)((a + b - 1) / b); };
  auto kpad = [](int K) { return (K + 31) & ~31; };

  // pack all weights to f16 padded
  auto packW = [&](const LayerP& L, int K, int Cout) {
    int Kp = kpad(K);
    _Float16* Wh = (_Float16*)AB((size_t)Cout * Kp * 2);
    long tot = (long)Cout * Kp;
    k_packW<<<cdiv(tot, 256), 256, 0, stream>>>(L.W, Wh, K, Kp, tot);
    return Wh;
  };
  _Float16* Wsa1[3] = { packW(sa1[0], 9, 64),    packW(sa1[1], 64, 64),   packW(sa1[2], 64, 128) };
  _Float16* Wsa2[3] = { packW(sa2[0], 131, 128), packW(sa2[1], 128, 128), packW(sa2[2], 128, 256) };
  _Float16* Wsa3[3] = { packW(sa3[0], 259, 256), packW(sa3[1], 256, 256), packW(sa3[2], 256, 512) };
  _Float16* Wfp3[2] = { packW(fp3[0], 768, 256), packW(fp3[1], 256, 256) };
  _Float16* Wfp2[2] = { packW(fp2[0], 384, 256), packW(fp2[1], 256, 128) };
  _Float16* Wfp1[2] = { packW(fp1[0], 134, 128), packW(fp1[1], 128, 128) };
  _Float16* Wc1     =   packW(cls1,   128, 128);
  _Float16* Wc2     =   packW(cls2,   128, 128);

  auto gemm = [&](const _Float16* Xp, _Float16* Wh, const LayerP& L, void* Yp,
                  int rows, int Kp, int Cout, int mode) {
    dim3 g(rows / 64, Cout / 16);   // all row counts are multiples of 64
    k_gemm_wmma<<<g, 128, 0, stream>>>(Xp, Wh, L.b, L.g, L.be, L.m, L.v,
                                       mode == 0 ? (_Float16*)Yp : nullptr,
                                       mode == 1 ? (float*)Yp : nullptr,
                                       Kp, Cout, mode);
  };

  // l0p
  k_build_l0p<<<cdiv((long)BB*NPTS*6, 256), 256, 0, stream>>>(feat, xyz, l0p, BB*NPTS*6);

  // ---- SA1: 16384 -> 1024, r=0.1, [9->64->64->128], group Kpad=32 ----
  k_fps<<<BB, 256, 0, stream>>>(xyz, mind, NPTS, 1024, sidx1);
  k_gather_xyz<<<cdiv(BB*1024, 256), 256, 0, stream>>>(xyz, sidx1, 1024, NPTS, nx1, BB*1024);
  k_knn<<<cdiv(BB*1024, 128), 128, 0, stream>>>(nx1, xyz, 1024, NPTS, 0.01f, knn1);
  long tg1 = (long)131072 * 32;
  k_group<<<cdiv(tg1, 256), 256, 0, stream>>>(xyz, l0p, nx1, knn1, 1024, NPTS, 6, 32, bufA, tg1);
  gemm(bufA, Wsa1[0], sa1[0], bufB, 131072, 32, 64, 0);
  gemm(bufB, Wsa1[1], sa1[1], bufA, 131072, 64, 64, 0);
  gemm(bufA, Wsa1[2], sa1[2], bufB, 131072, 64, 128, 0);
  long tp1 = (long)BB * 1024 * 128;
  k_maxpool<<<cdiv(tp1, 256), 256, 0, stream>>>(bufB, 128, l1p, tp1);

  // ---- SA2: 1024 -> 256, r=0.2, [131->128->128->256], group Kpad=160 ----
  k_fps<<<BB, 256, 0, stream>>>(nx1, mind, 1024, 256, sidx2);
  k_gather_xyz<<<cdiv(BB*256, 256), 256, 0, stream>>>(nx1, sidx2, 256, 1024, nx2, BB*256);
  k_knn<<<cdiv(BB*256, 128), 128, 0, stream>>>(nx2, nx1, 256, 1024, 0.04f, knn2);
  long tg2 = (long)32768 * 160;
  k_group<<<cdiv(tg2, 256), 256, 0, stream>>>(nx1, l1p, nx2, knn2, 256, 1024, 128, 160, bufA, tg2);
  gemm(bufA, Wsa2[0], sa2[0], bufB, 32768, 160, 128, 0);
  gemm(bufB, Wsa2[1], sa2[1], bufA, 32768, 128, 128, 0);
  gemm(bufA, Wsa2[2], sa2[2], bufB, 32768, 128, 256, 0);
  long tp2 = (long)BB * 256 * 256;
  k_maxpool<<<cdiv(tp2, 256), 256, 0, stream>>>(bufB, 256, l2p, tp2);

  // ---- SA3: 256 -> 64, r=0.4, [259->256->256->512], group Kpad=288 ----
  k_fps<<<BB, 256, 0, stream>>>(nx2, mind, 256, 64, sidx3);
  k_gather_xyz<<<cdiv(BB*64, 256), 256, 0, stream>>>(nx2, sidx3, 64, 256, nx3, BB*64);
  k_knn<<<cdiv(BB*64, 128), 128, 0, stream>>>(nx3, nx2, 64, 256, 0.16f, knn3);
  long tg3 = (long)8192 * 288;
  k_group<<<cdiv(tg3, 256), 256, 0, stream>>>(nx2, l2p, nx3, knn3, 64, 256, 256, 288, bufA, tg3);
  gemm(bufA, Wsa3[0], sa3[0], bufB, 8192, 288, 256, 0);
  gemm(bufB, Wsa3[1], sa3[1], bufA, 8192, 256, 256, 0);
  gemm(bufA, Wsa3[2], sa3[2], bufB, 8192, 256, 512, 0);
  long tp3 = (long)BB * 64 * 512;
  k_maxpool<<<cdiv(tp3, 256), 256, 0, stream>>>(bufB, 512, l3p, tp3);

  // ---- FP3: interp l3 -> l2, concat l2p, [768->256->256] ----
  k_fp_interp<<<cdiv(BB*256, 128), 128, 0, stream>>>(nx2, nx3, l3p, l2p, 256, 64, 512, 256, 768, bufA);
  gemm(bufA, Wfp3[0], fp3[0], bufB, 1024, 768, 256, 0);
  gemm(bufB, Wfp3[1], fp3[1], fp3o, 1024, 256, 256, 0);

  // ---- FP2: interp l2 -> l1, concat l1p, [384->256->128] ----
  k_fp_interp<<<cdiv(BB*1024, 128), 128, 0, stream>>>(nx1, nx2, fp3o, l1p, 1024, 256, 256, 128, 384, bufA);
  gemm(bufA, Wfp2[0], fp2[0], bufB, 4096, 384, 256, 0);
  gemm(bufB, Wfp2[1], fp2[1], fp2o, 4096, 256, 128, 0);

  // ---- FP1: interp l1 -> l0, concat l0p, [134->128->128], Kpad=160 ----
  k_fp_interp<<<cdiv(BB*NPTS, 128), 128, 0, stream>>>(xyz, nx1, fp2o, l0p, NPTS, 1024, 128, 6, 160, bufA);
  gemm(bufA, Wfp1[0], fp1[0], bufB, 65536, 160, 128, 0);
  gemm(bufB, Wfp1[1], fp1[1], bufA, 65536, 128, 128, 0);

  // ---- head: cls1 (BN+ReLU), cls2 (bias -> f32), L2 normalize ----
  gemm(bufA, Wc1, cls1, bufB, 65536, 128, 128, 0);
  gemm(bufB, Wc2, cls2, clsf, 65536, 128, 128, 1);
  k_norm_out<<<cdiv(BB*NPTS, 256), 256, 0, stream>>>(clsf, (float*)d_out, NPTS);
}
